// VectorQuantizer_38319698215386
// MI455X (gfx1250) — matrix operand order repaired and checked by reading, not compiled, not run on hardware
//
#include <hip/hip_runtime.h>

// ---------------- problem constants ----------------
#define BQ   8
#define CQ   256
#define LQ   2048
#define KQ   8192
#define MQ   (BQ * LQ)            // 16384 rows
#define NELEM (BQ * CQ * LQ)      // 4194304
#define NSPLIT 2                  // N dimension split across blockIdx.y
#define NTILES_SPLIT (KQ / 16 / NSPLIT)   // 256 tiles per split
#define LDS_ROW 264               // 256 + 8 u16 pad (16B) -> conflict-free b128

typedef __attribute__((ext_vector_type(16))) __bf16 v16bf;
typedef __attribute__((ext_vector_type(8)))  float  v8f;
typedef __attribute__((ext_vector_type(4)))  unsigned int u32x4;

union FragBF {
  v16bf v;
  u32x4 q[2];
};

__device__ __forceinline__ unsigned short f32_to_bf16(float f) {
  union { float f; unsigned int u; } cvt; cvt.f = f;
  unsigned int u = cvt.u;
  unsigned int r = u + 0x7FFFu + ((u >> 16) & 1u);   // round-to-nearest-even
  return (unsigned short)(r >> 16);
}

// ---------- kernel 1: codebook -> bf16, csq[k] = ||codebook[k]||^2 ----------
__global__ void vq_prep_codebook(const float* __restrict__ cb,
                                 unsigned short* __restrict__ cb_bf,
                                 float* __restrict__ csq) {
  int k = blockIdx.x;
  int t = threadIdx.x;                       // 256 threads, one per channel
  float v = cb[k * CQ + t];
  cb_bf[k * CQ + t] = f32_to_bf16(v);
  __shared__ float red[256];
  red[t] = v * v;
  __syncthreads();
  for (int s = 128; s > 0; s >>= 1) {
    if (t < s) red[t] += red[t + s];
    __syncthreads();
  }
  if (t == 0) csq[k] = red[0];
}

// ---------- kernel 2: x (B,C,L) -> xt_bf16 (B*L, C), tiled transpose ----------
__global__ void vq_prep_x(const float* __restrict__ x,
                          unsigned short* __restrict__ xt_bf) {
  __shared__ float tile[32][33];
  int b  = blockIdx.z;
  int l0 = blockIdx.x * 32;
  int c0 = blockIdx.y * 32;
  int tx = threadIdx.x, ty = threadIdx.y;    // 32 x 8
  #pragma unroll
  for (int i = 0; i < 32; i += 8)
    tile[ty + i][tx] = x[(size_t)b * CQ * LQ + (size_t)(c0 + ty + i) * LQ + (l0 + tx)];
  __syncthreads();
  #pragma unroll
  for (int i = 0; i < 32; i += 8)
    xt_bf[(size_t)(b * LQ + l0 + ty + i) * CQ + (c0 + tx)] = f32_to_bf16(tile[tx][ty + i]);
}

// ---------- kernel 3: WMMA GEMM + per-split argmax(csq - 2*dot) ----------
// Block: 256 threads = 8 waves. Each wave owns 32 M-rows (two 16-row sub-tiles
// -> two independent WMMA accumulator chains sharing each B fragment).
// Block covers 256 M-rows; blockIdx.y splits the 8192 codebook rows in 2.
__launch_bounds__(256)
__global__ void vq_gemm_argmax(const unsigned short* __restrict__ xt_bf,
                               const unsigned short* __restrict__ cb_bf,
                               const float* __restrict__ csq,
                               float* __restrict__ bestValH,   // [NSPLIT][MQ]
                               int* __restrict__ bestIdxH) {   // [NSPLIT][MQ]
  __shared__ __align__(16) unsigned short ldsB[2][16 * LDS_ROW];

  const int tid  = threadIdx.x;
  const int wave = tid >> 5;
  const int lane = tid & 31;
  const int hsel = lane >> 4;                 // wave32 half select
  const int l16  = lane & 15;
  const int split = blockIdx.y;
  const int waveRow = blockIdx.x * 256 + wave * 32;
  const int tile0 = split * NTILES_SPLIT;

  // --- A fragments: two 16x256 bf16 sub-tiles, rows = waveRow + sub*16 + l16 ---
  FragBF a0[8], a1[8];
  const unsigned short* arow0 = xt_bf + (size_t)(waveRow + l16) * CQ;
  const unsigned short* arow1 = xt_bf + (size_t)(waveRow + 16 + l16) * CQ;
  #pragma unroll
  for (int f = 0; f < 8; ++f) {
    int kc = f * 32;
    a0[f].q[0] = *(const u32x4*)(arow0 + kc + hsel * 8);
    a0[f].q[1] = *(const u32x4*)(arow0 + kc + 16 + hsel * 8);
    a1[f].q[0] = *(const u32x4*)(arow1 + kc + hsel * 8);
    a1[f].q[1] = *(const u32x4*)(arow1 + kc + 16 + hsel * 8);
  }

  float bestVal[16];
  int   bestIdx[16];
  #pragma unroll
  for (int j = 0; j < 16; ++j) { bestVal[j] = -3.4e38f; bestIdx[j] = 0; }

  // cooperative codebook-tile loader: 16 rows x 256 cols bf16 (8KB), padded LDS
  auto loadTile = [&](int t, int bufNum) {
    const unsigned short* src = cb_bf + (size_t)(tile0 + t) * 16 * CQ + tid * 16;
    u32x4 q0 = *(const u32x4*)(src);
    u32x4 q1 = *(const u32x4*)(src + 8);
    int row = tid >> 4;
    int col = (tid & 15) << 4;
    unsigned short* dst = &ldsB[bufNum][row * LDS_ROW + col];
    *(u32x4*)(dst)     = q0;
    *(u32x4*)(dst + 8) = q1;
  };

  loadTile(0, 0);
  __syncthreads();

  for (int t = 0; t < NTILES_SPLIT; ++t) {
    int buf = t & 1;
    if (t + 1 < NTILES_SPLIT) loadTile(t + 1, buf ^ 1);   // prefetch other buffer

    int nBase = (tile0 + t) * 16;
    float cs = csq[nBase + l16];

    // B fragment base: N = lane&15, K = hsel*8 .. +7 and 16 + hsel*8 .. +7 (table order)
    const unsigned short* brow = &ldsB[buf][l16 * LDS_ROW + hsel * 8];

    v8f acc0 = {0.f, 0.f, 0.f, 0.f, 0.f, 0.f, 0.f, 0.f};
    v8f acc1 = {0.f, 0.f, 0.f, 0.f, 0.f, 0.f, 0.f, 0.f};
    #pragma unroll
    for (int f = 0; f < 8; ++f) {
      FragBF bf;
      bf.q[0] = *(const u32x4*)(brow + f * 32);
      bf.q[1] = *(const u32x4*)(brow + f * 32 + 16);
      // two independent accumulation chains sharing one B fragment
      acc0 = __builtin_amdgcn_wmma_f32_16x16x32_bf16(
          false, a0[f].v, false, bf.v, (short)0, acc0, false, false);
      acc1 = __builtin_amdgcn_wmma_f32_16x16x32_bf16(
          false, a1[f].v, false, bf.v, (short)0, acc1, false, false);
    }

    int n = nBase + l16;
    #pragma unroll
    for (int j = 0; j < 8; ++j) {
      float s0 = cs - 2.0f * acc0[j];        // sqrt monotone; x_sq constant per row
      float s1 = cs - 2.0f * acc1[j];
      if (s0 > bestVal[j])     { bestVal[j]     = s0; bestIdx[j]     = n; }
      if (s1 > bestVal[j + 8]) { bestVal[j + 8] = s1; bestIdx[j + 8] = n; }
    }
    __syncthreads();                          // release buf for next prefetch
  }

  // --- per-row argmax across 16 columns (lanes within each 16-half) ---
  #pragma unroll
  for (int sub = 0; sub < 2; ++sub) {
    #pragma unroll
    for (int j = 0; j < 8; ++j) {
      float v = bestVal[sub * 8 + j];
      int   ix = bestIdx[sub * 8 + j];
      #pragma unroll
      for (int m = 8; m >= 1; m >>= 1) {
        float ov = __shfl_xor(v, m, 32);
        int   oi = __shfl_xor(ix, m, 32);
        if (ov > v || (ov == v && oi < ix)) { v = ov; ix = oi; }  // first-index ties
      }
      if (l16 == 0) {                         // lanes 0 & 16: rows j and j+8
        int mg = waveRow + sub * 16 + j + hsel * 8;
        bestValH[split * MQ + mg] = v;
        bestIdxH[split * MQ + mg] = ix;
      }
    }
  }
}

// ---------- kernel 3b: combine the two N-splits (deterministic) ----------
__global__ void vq_combine(const float* __restrict__ bestValH,
                           const int* __restrict__ bestIdxH,
                           int* __restrict__ codeIdx,
                           float* __restrict__ codeOut) {
  int m = blockIdx.x * 256 + threadIdx.x;
  float v0 = bestValH[m];        int i0 = bestIdxH[m];
  float v1 = bestValH[MQ + m];   int i1 = bestIdxH[MQ + m];
  // split 0 indices are all lower -> ties go to split 0 (jnp.argmax first-index)
  int ix = (v1 > v0) ? i1 : i0;
  codeIdx[m] = ix;
  codeOut[m] = (float)ix;
}

// ---------- kernel 4: gather emb + per-block loss partials ----------
__global__ void vq_gather_loss(const float* __restrict__ x,
                               const float* __restrict__ cb,
                               const int* __restrict__ codeIdx,
                               float* __restrict__ embOut,
                               float* __restrict__ partials) {
  int idx = blockIdx.x * 256 + threadIdx.x;  // over B*C*L, power of two
  int l = idx & (LQ - 1);
  int c = (idx >> 11) & (CQ - 1);
  int b = idx >> 19;
  int code = codeIdx[b * LQ + l];
  float e = cb[(size_t)code * CQ + c];
  embOut[idx] = e;                           // emb[b,c,l] = codebook[code[b,l], c]
  float d = x[idx] - e;
  __shared__ float red[256];
  red[threadIdx.x] = d * d;
  __syncthreads();
  for (int s = 128; s > 0; s >>= 1) {
    if (threadIdx.x < s) red[threadIdx.x] += red[threadIdx.x + s];
    __syncthreads();
  }
  if (threadIdx.x == 0) partials[blockIdx.x] = red[0];
}

// ---------- kernel 5: deterministic final loss reduction ----------
__global__ void vq_final(const float* __restrict__ partials, int n,
                         float* __restrict__ lossOut) {
  __shared__ float red[256];
  float s = 0.f;
  for (int i = threadIdx.x; i < n; i += 256) s += partials[i];
  red[threadIdx.x] = s;
  __syncthreads();
  for (int st = 128; st > 0; st >>= 1) {
    if (threadIdx.x < st) red[threadIdx.x] += red[threadIdx.x + st];
    __syncthreads();
  }
  if (threadIdx.x == 0) *lossOut = red[0] / (float)NELEM;
}

extern "C" void kernel_launch(void* const* d_in, const int* in_sizes, int n_in,
                              void* d_out, int out_size, void* d_ws, size_t ws_size,
                              hipStream_t stream) {
  const float* x  = (const float*)d_in[0];   // (8,256,2048)
  const float* cb = (const float*)d_in[1];   // (8192,256)

  float* out     = (float*)d_out;
  float* codeOut = out;                      // 16384 (code as float)
  float* embOut  = out + MQ;                 // 4194304
  float* lossOut = out + MQ + NELEM;         // 1

  char* ws = (char*)d_ws;
  unsigned short* xt_bf = (unsigned short*)ws;                          // 8 MiB
  unsigned short* cb_bf = (unsigned short*)(ws + (8u << 20));           // 4 MiB
  float* csq            = (float*)(ws + (12u << 20));                   // 32 KiB
  int*   codeIdx        = (int*)  (ws + (12u << 20) + (64u  << 10));    // 64 KiB
  float* partials       = (float*)(ws + (12u << 20) + (192u << 10));    // 64 KiB
  float* bestValH       = (float*)(ws + (12u << 20) + (256u << 10));    // 128 KiB
  int*   bestIdxH       = (int*)  (ws + (12u << 20) + (384u << 10));    // 128 KiB

  vq_prep_codebook<<<KQ, 256, 0, stream>>>(cb, cb_bf, csq);
  dim3 tgrid(LQ / 32, CQ / 32, BQ);
  vq_prep_x<<<tgrid, dim3(32, 8, 1), 0, stream>>>(x, xt_bf);
  dim3 ggrid(MQ / 256, NSPLIT, 1);
  vq_gemm_argmax<<<ggrid, 256, 0, stream>>>(xt_bf, cb_bf, csq, bestValH, bestIdxH);
  vq_combine<<<MQ / 256, 256, 0, stream>>>(bestValH, bestIdxH, codeIdx, codeOut);
  vq_gather_loss<<<NELEM / 256, 256, 0, stream>>>(x, cb, codeIdx, embOut, partials);
  vq_final<<<1, 256, 0, stream>>>(partials, NELEM / 256, lossOut);
}
